// ARC_34342558498816
// MI455X (gfx1250) — matrix operand order repaired and checked by reading, not compiled, not run on hardware
//
#include <hip/hip_runtime.h>

typedef __attribute__((ext_vector_type(16))) __bf16 v16bf;
typedef __attribute__((ext_vector_type(8)))  float  v8f;

// Problem constants (match reference)
#define BB     8
#define TT     2304          // SEQ + 2*STATE
#define STATE_ 128
#define MIDE   2176          // STATE + SEQ
#define DD     1024
#define DKV    64
#define NTILE  144           // TT/16

static __device__ __forceinline__ v8f wmma_bf16(v16bf a, v16bf b, v8f c) {
  // D = A(16x32 bf16) * B(32x16 bf16) + C(16x16 f32)
  return __builtin_amdgcn_wmma_f32_16x16x32_bf16(false, a, false, b, (short)0, c, false, false);
}
static __device__ __forceinline__ v16bf loadv16(const __bf16* p) {
  return *(const v16bf*)p;   // 32B contiguous, alignment guaranteed by layout
}

// ---------------------------------------------------------------------------
// Kernel 0a: transpose cope_emb (f32 [64][T]) -> copeT bf16 [T][64]
// ---------------------------------------------------------------------------
__global__ void k0_copeT(const float* __restrict__ cope, __bf16* __restrict__ ct) {
  int i = blockIdx.x * 256 + threadIdx.x;       // i < T*64
  int t = i >> 6, d = i & 63;
  ct[i] = (__bf16)cope[d * TT + t];
}

// ---------------------------------------------------------------------------
// Kernel 0b: generic f32 -> bf16 convert (used for the 9 weight matrices so
// the k1 GEMM B-operand is a single contiguous 32B bf16 load, no per-block
// v_cvt_pk_bf16_f32 in the K-loop)
// ---------------------------------------------------------------------------
__global__ void k0_cvt(const float* __restrict__ src, __bf16* __restrict__ dst) {
  int i = blockIdx.x * 256 + threadIdx.x;
  dst[i] = (__bf16)src[i];
}

// ---------------------------------------------------------------------------
// Kernel 1: segmented LayerNorm + QKV projection via WMMA.
// One block = one 16-row tile (tiles never straddle segment bounds: 128, 2176
// are multiples of 16). 512 threads = 16 waves: wave w normalizes row w into
// LDS bf16; then waves 0..11 each compute one 16-col output tile (q:0-3,
// k:4-7, v:8-11) with a K=1024 WMMA loop over pre-converted bf16 weights.
// ---------------------------------------------------------------------------
__global__ void k1_ln_qkv(const float* __restrict__ x,
    const float* ln_w, const float* ln_b,
    const float* ln_ss_w, const float* ln_ss_b,
    const float* ln_se_w, const float* ln_se_b,
    const __bf16* __restrict__ wb,               // 9 x [64][1024] bf16: q,k,v | *_ss | *_se
    __bf16* __restrict__ qb, __bf16* __restrict__ kb, __bf16* __restrict__ vT) {
  __shared__ __bf16 xn[16 * DD];                 // 32KB normalized tile (A-source)
  const int b = blockIdx.x / NTILE, tile = blockIdx.x % NTILE, t0 = tile * 16;
  const int w = threadIdx.x >> 5, lane = threadIdx.x & 31;

  const float *LW, *LB;
  int wsel;                                      // which weight triple
  if (t0 < STATE_)     { LW = ln_ss_w; LB = ln_ss_b; wsel = 1; }
  else if (t0 < MIDE)  { LW = ln_w;    LB = ln_b;    wsel = 0; }
  else                 { LW = ln_se_w; LB = ln_se_b; wsel = 2; }

  { // ---- LayerNorm: wave w owns row w; 32 lanes x 32 contiguous elems ----
    const float* xr = x + ((size_t)b * TT + t0 + w) * DD;
    float vals[32], s = 0.f, ss = 0.f;
    #pragma unroll
    for (int j = 0; j < 32; ++j) { float v = xr[lane * 32 + j]; vals[j] = v; s += v; ss += v * v; }
    for (int off = 16; off > 0; off >>= 1) { s += __shfl_xor(s, off, 32); ss += __shfl_xor(ss, off, 32); }
    const float mu = s * (1.0f / DD);
    const float rstd = rsqrtf(ss * (1.0f / DD) - mu * mu + 1e-5f);
    #pragma unroll
    for (int j = 0; j < 32; ++j) {
      int d = lane * 32 + j;
      xn[w * DD + d] = (__bf16)((vals[j] - mu) * rstd * LW[d] + LB[d]);
    }
  }
  __syncthreads();

  if (w < 12) {                                  // uniform branch: EXEC all-ones inside
    const int grp = w >> 2, sub = w & 3;         // grp 0=q 1=k 2=v
    const int lo = lane & 15, hi = lane >> 4;
    // weight matrix: order in wb is wq,wk,wv, wq_ss,wk_ss,wv_ss, wq_se,wk_se,wv_se
    const __bf16* W = wb + ((size_t)(wsel * 3 + grp)) * (DKV * DD);
    const int col = sub * 16 + lo;               // output column (B-operand N = lane)
    const __bf16* Wrow = W + (size_t)col * DD;
    v8f acc = {};
    for (int k0 = 0; k0 < DD; k0 += 32) {
      v16bf a;
      #pragma unroll
      for (int j = 0; j < 8; ++j) {              // A 16x32: row=lo, K = hi*16+j
        a[j]     = xn[lo * DD + k0 + hi * 16 + j];
        a[8 + j] = xn[lo * DD + k0 + hi * 16 + 8 + j];
      }
      v16bf bf = loadv16(Wrow + k0 + hi * 16);   // B 32x16: N=lo fixed, K = hi*16+j
      acc = wmma_bf16(a, bf, acc);
    }
    if (grp < 2) {                               // q,k: row-major bf16 [B*T][64]
      __bf16* dst = (grp == 0) ? qb : kb;
      #pragma unroll
      for (int j = 0; j < 8; ++j) {
        int M = hi * 8 + j;
        dst[((size_t)b * TT + t0 + M) * DKV + sub * 16 + lo] = (__bf16)acc[j];
      }
    } else {                                     // v: transposed bf16 [B][64][T]
      #pragma unroll
      for (int j = 0; j < 8; ++j) {
        int M = hi * 8 + j;
        vT[(size_t)b * DKV * TT + (size_t)(sub * 16 + lo) * TT + t0 + M] = (__bf16)acc[j];
      }
    }
  }
}

// ---------------------------------------------------------------------------
// Kernel 2: CoPE attention for one (b, 16-row q-tile). 256 threads = 8 waves.
// LDS (dynamic, 290KB — needs CDNA5's 320KB/WGP):
//   lgt [16][2304] f32  : logits -> scores -> exp()
//   lint[16][2304] f32  : q@cope, then per-row overlaid with bf16 probs
//   qs  [16][64]   bf16 : q tile (WMMA A-source)
// Phase 1: WMMA q@k^T and q@copeT (288 16x16 tile-jobs over 8 waves), with
//          global_prefetch of the next tile to hide HBM latency behind WMMA.
// Phase 2: wave w owns rows 2w,2w+1: sigmoid + wave32 suffix-scan reverse
//          cumsum, bias gather/interp, causal mask, 2-pass softmax, bf16 probs
// Phase 3: WMMA probs@v with split-K across wave pairs + LDS reduce
// ---------------------------------------------------------------------------
__global__ void k2_attn(const __bf16* __restrict__ qb, const __bf16* __restrict__ kb,
                        const __bf16* __restrict__ vT, const __bf16* __restrict__ ct,
                        float* __restrict__ out) {
  extern __shared__ char smem[];
  float*  lgt  = (float*)smem;
  float*  lint = lgt + 16 * TT;
  __bf16* qs   = (__bf16*)(lint + 16 * TT);

  const int b = blockIdx.x / NTILE, tile = blockIdx.x % NTILE, t0 = tile * 16;
  const int tid = threadIdx.x, w = tid >> 5, lane = tid & 31;
  const int lo = lane & 15, hi = lane >> 4;

  for (int i = tid; i < 16 * DKV; i += 256)      // q tile -> LDS
    qs[i] = qb[((size_t)b * TT + t0 + (i >> 6)) * DKV + (i & 63)];
  __syncthreads();

  v16bf a0, a1;                                  // q A-fragments (K=0..31, 32..63)
  #pragma unroll
  for (int j = 0; j < 8; ++j) {
    a0[j]     = qs[lo * DKV +      hi * 16 + j];
    a0[8 + j] = qs[lo * DKV +  8 + hi * 16 + j];
    a1[j]     = qs[lo * DKV + 32 + hi * 16 + j];
    a1[8 + j] = qs[lo * DKV + 40 + hi * 16 + j];
  }

  // ---- Phase 1: logits and logits_int, 16x16 tiles ----
  const __bf16* kbase = kb + (size_t)b * TT * DKV;
  for (int job = w; job < 2 * NTILE; job += 8) {
    const int  ttile = job % NTILE;
    const bool isK   = job < NTILE;
    const __bf16* src = isK ? (kbase + (size_t)(ttile * 16 + lo) * DKV)
                            : (ct + (size_t)(ttile * 16 + lo) * DKV);
    { // prefetch next job's tile (stride-8 walk => address known)
      const int jn = job + 8;
      if (jn < 2 * NTILE) {
        const int  tn  = jn % NTILE;
        const __bf16* pn = (jn < NTILE) ? (kbase + (size_t)(tn * 16 + lo) * DKV)
                                        : (ct + (size_t)(tn * 16 + lo) * DKV);
        __builtin_prefetch(pn, 0, 1);            // -> global_prefetch_b8
      }
    }
    v16bf b0 = loadv16(src + hi * 16);
    v16bf b1 = loadv16(src + 32 + hi * 16);
    v8f acc = {};
    acc = wmma_bf16(a0, b0, acc);
    acc = wmma_bf16(a1, b1, acc);
    float* dst = isK ? lgt : lint;
    #pragma unroll
    for (int j = 0; j < 8; ++j)
      dst[(hi * 8 + j) * TT + ttile * 16 + lo] = acc[j];
  }
  __syncthreads();

  // ---- Phase 2: gates, reverse cumsum (wave suffix scan), bias, softmax ----
  const int CH = TT / 32;                        // 72 elems per lane
  for (int rr = 0; rr < 2; ++rr) {
    const int r = w * 2 + rr, qg = t0 + r;
    float* L  = lgt  + r * TT;
    float* LI = lint + r * TT;
    float lsum = 0.f;
    for (int j = 0; j < CH; ++j)
      lsum += 1.0f / (1.0f + __expf(-L[lane * CH + j]));
    float inc = lsum;                            // inclusive suffix across lanes
    for (int off = 1; off < 32; off <<= 1) {
      float t = __shfl_down(inc, off, 32);
      if (lane + off < 32) inc += t;
    }
    float running = inc - lsum;                  // sum over lanes > me
    float m = -3.0e38f;
    for (int j = CH - 1; j >= 0; --j) {          // descending: pos[k] = sum_{j>=k} g
      const int k = lane * CH + j;
      const float lg = L[k];
      running += 1.0f / (1.0f + __expf(-lg));
      const float pos = fminf(running, (float)(TT - 1));
      const float pf  = floorf(pos);
      const int   ic  = (int)ceilf(pos), ifl = (int)pf;
      const float wt  = pos - pf;
      const float bias = LI[ic] * wt + LI[ifl] * (1.0f - wt);
      float sc = lg * 0.125f + bias;             // 1/sqrt(64)
      sc = (k <= qg) ? sc : -3.0e38f;            // causal
      L[k] = sc;
      m = fmaxf(m, sc);
    }
    for (int off = 16; off > 0; off >>= 1) m = fmaxf(m, __shfl_xor(m, off, 32));
    float sum = 0.f;
    for (int j = 0; j < CH; ++j) {
      const int k = lane * CH + j;
      const float e = __expf(L[k] - m);
      L[k] = e; sum += e;
    }
    for (int off = 16; off > 0; off >>= 1) sum += __shfl_xor(sum, off, 32);
    const float invs = 1.0f / sum;
    __bf16* pr = (__bf16*)LI;                    // overlay probs on own lint row
    for (int j = 0; j < CH; ++j) {
      const int k = lane * CH + j;
      pr[k] = (__bf16)(L[k] * invs);
    }
  }
  __syncthreads();

  // ---- Phase 3: out = probs @ v, split-K over wave pairs ----
  const int n = w & 3, half = w >> 2;            // n: 16-col d-tile, half: K half
  v8f acc = {};
  const __bf16* vrow = vT + (size_t)b * DKV * TT + (size_t)(n * 16 + lo) * TT;
  for (int kt = half * 36; kt < half * 36 + 36; ++kt) {
    const int k0 = kt * 32;
    v16bf a, bf;
    const __bf16* prlo = (const __bf16*)(lint + lo * TT);
    #pragma unroll
    for (int j = 0; j < 8; ++j) {
      a[j]     = prlo[k0 +     hi * 16 + j];
      a[8 + j] = prlo[k0 + 8 + hi * 16 + j];
    }
    bf = loadv16(vrow + k0 + hi * 16);
    acc = wmma_bf16(a, bf, acc);
  }
  float* scr = lgt;                              // reuse: 4 tiles * 256 f32
  if (half == 1) {
    #pragma unroll
    for (int j = 0; j < 8; ++j)
      scr[n * 256 + (hi * 8 + j) * 16 + lo] = acc[j];
  }
  __syncthreads();
  if (half == 0) {
    #pragma unroll
    for (int j = 0; j < 8; ++j) {
      const int M = hi * 8 + j;
      out[((size_t)b * TT + t0 + M) * DKV + n * 16 + lo] =
          acc[j] + scr[n * 256 + M * 16 + lo];
    }
  }
}

// ---------------------------------------------------------------------------
extern "C" void kernel_launch(void* const* d_in, const int* in_sizes, int n_in,
                              void* d_out, int out_size, void* d_ws, size_t ws_size,
                              hipStream_t stream) {
  const float* x       = (const float*)d_in[0];
  const float* ln_w    = (const float*)d_in[1];
  const float* ln_b    = (const float*)d_in[2];
  const float* ln_ss_w = (const float*)d_in[3];
  const float* ln_ss_b = (const float*)d_in[4];
  const float* ln_se_w = (const float*)d_in[5];
  const float* ln_se_b = (const float*)d_in[6];
  const float* cope    = (const float*)d_in[16];

  const size_t nqkv = (size_t)BB * TT * DKV;     // bf16 elems each
  const size_t nw   = (size_t)DKV * DD;          // elems per weight matrix
  __bf16* qb = (__bf16*)d_ws;
  __bf16* kb = qb + nqkv;
  __bf16* vT = kb + nqkv;
  __bf16* ct = vT + nqkv;                        // [T][64] bf16
  __bf16* wb = ct + (size_t)TT * DKV;            // 9 x [64][1024] bf16

  // bf16 weights, ordered q,k,v | q_ss,k_ss,v_ss | q_se,k_se,v_se
  // d_in indices:                7,8,9 | 10,11,12 | 13,14,15  (same order)
  for (int m = 0; m < 9; ++m)
    k0_cvt<<<(int)(nw / 256), 256, 0, stream>>>((const float*)d_in[7 + m], wb + m * nw);
  k0_copeT<<<(TT * DKV) / 256, 256, 0, stream>>>(cope, ct);

  k1_ln_qkv<<<BB * NTILE, 512, 0, stream>>>(x,
      ln_w, ln_b, ln_ss_w, ln_ss_b, ln_se_w, ln_se_b, wb, qb, kb, vT);

  const int smem = 16 * TT * 4 * 2 + 16 * DKV * 2;   // 296960 B (fits 320KB WGP LDS)
  k2_attn<<<BB * NTILE, 256, smem, stream>>>(qb, kb, vT, ct, (float*)d_out);
}
